// SplatGaussian2D_38474317038266
// MI455X (gfx1250) — matrix operand order repaired and checked by reading, not compiled, not run on hardware
//
#include <hip/hip_runtime.h>

#define GN       8192
#define NSH      4
#define IMG_H    1024
#define IMG_W    1024
#define S_MIN_C  (1.0f/30.0f)
#define S_MAX_C  (1.0f/0.75f)
#define MU_BRD   1.05f
#define EPS_C    1e-6f
#define TILE     16
#define TX_TILES (IMG_W/TILE)   // 64
#define TY_TILES (IMG_H/TILE)   // 64
#define REC_F    16             // floats per gaussian record (64 B)

typedef __attribute__((ext_vector_type(16))) _Float16 v16h;
typedef __attribute__((ext_vector_type(8)))  float    v8f;
typedef __attribute__((ext_vector_type(4)))  unsigned u32x4;
typedef __attribute__((ext_vector_type(4)))  int      i32x4;
typedef __attribute__((ext_vector_type(8)))  int      i32x8;

#if defined(__has_builtin)
#if __has_builtin(__builtin_amdgcn_tensor_load_to_lds) && \
    __has_builtin(__builtin_amdgcn_s_wait_tensorcnt)
#define USE_TDM 1
#endif
#endif
#ifndef USE_TDM
#define USE_TDM 0
#endif

// Record layout (16 floats):
// [0]=xlo [1]=xhi [2]=ylo [3]=yhi   (clipped pixel bbox, inclusive)
// [4]=mu_px [5]=mu_py [6]=qa [7]=qb (q = qa*dx^2 + qb*dx*dy + qc*dy^2)
// [8]=qc [9]=opacity [10]=R [11]=G [12]=B [13]=1.0 (weight col) [14..15]=0

__global__ __launch_bounds__(256) void prep_kernel(
    const float* __restrict__ xyz,  const float* __restrict__ feat,
    const float* __restrict__ opac, const float* __restrict__ scal,
    const float* __restrict__ rot,  float* __restrict__ grec)
{
  int g = blockIdx.x * 256 + threadIdx.x;
  if (g >= GN) return;

  float mx = fminf(fmaxf(xyz[2*g+0], -1.f), 1.f) * MU_BRD;
  float my = fminf(fmaxf(xyz[2*g+1], -1.f), 1.f) * MU_BRD;
  float sx = fminf(fmaxf(scal[2*g+0], 0.f), 1.f) * (S_MAX_C - S_MIN_C) + S_MIN_C;
  float sy = fminf(fmaxf(scal[2*g+1], 0.f), 1.f) * (S_MAX_C - S_MIN_C) + S_MIN_C;
  float r0 = rot[g];
  float ang = (r0 - floorf(r0)) * 6.2831853071795864f;
  float op  = fminf(fmaxf(opac[g], 0.f), 1.f);

  float s1, c1;
  __sincosf(ang, &s1, &c1);

  // circular harmonic basis [1, sin k*t, cos k*t] via angle-addition recurrence
  float bs[2*NSH+1];
  bs[0] = 1.f;
  float sk = s1, ck = c1;
  bs[1] = sk; bs[2] = ck;
  #pragma unroll
  for (int k = 1; k < NSH; ++k) {
    float sn = sk*c1 + ck*s1;
    float cn = ck*c1 - sk*s1;
    sk = sn; ck = cn;
    bs[2*k+1] = sk; bs[2*k+2] = ck;
  }
  float R = 0.f, G = 0.f, B = 0.f;
  const float* f = feat + (size_t)g * (2*NSH+1) * 3;
  #pragma unroll
  for (int m = 0; m < 2*NSH+1; ++m) {
    R = fmaf(bs[m], f[3*m+0], R);
    G = fmaf(bs[m], f[3*m+1], G);
    B = fmaf(bs[m], f[3*m+2], B);
  }

  float mupx = (mx + 1.f) * 0.5f * (float)(IMG_W - 1);
  float mupy = (my + 1.f) * 0.5f * (float)(IMG_H - 1);
  float bx = floorf(mupx), by = floorf(mupy);
  float xlo = fmaxf(bx - 24.f, 0.f), xhi = fminf(bx + 23.f, (float)(IMG_W-1));
  float ylo = fmaxf(by - 24.f, 0.f), yhi = fminf(by + 23.f, (float)(IMG_H-1));

  // v = diag(S) R^T d ; |v|^2 = qa dx^2 + qb dx dy + qc dy^2
  float sx2 = sx*sx, sy2 = sy*sy;
  float qa = sx2*c1*c1 + sy2*s1*s1;
  float qb = 2.f*c1*s1*(sx2 - sy2);
  float qc = sx2*s1*s1 + sy2*c1*c1;

  float4* o = (float4*)(grec + (size_t)g * REC_F);
  o[0] = make_float4(xlo, xhi, ylo, yhi);
  o[1] = make_float4(mupx, mupy, qa, qb);
  o[2] = make_float4(qc, op, R, G);
  o[3] = make_float4(B, 1.f, 0.f, 0.f);
}

#if USE_TDM
// One TDM gather: 16 rows of 64B (8 x 8B elements) from grec, row ids from idx[],
// packed 2 u16 per dword, landing contiguously at LDS byte offset lds_off.
// D# per CDNA5 ISA (gather_mode=1, 16-bit indices).
__device__ __forceinline__ void tdm_gather16(unsigned lds_off,
                                             const float* __restrict__ g,
                                             const unsigned* idx)
{
  unsigned long long ga = (unsigned long long)(const void*)g;
  u32x4 g0;
  g0.x = 0x80000001u;                                   // count=1 | gather_mode<<31
  g0.y = lds_off;                                       // lds_addr
  g0.z = (unsigned)ga;                                  // global_addr[31:0]
  g0.w = ((unsigned)(ga >> 32) & 0x01FFFFFFu) | (2u << 30); // addr[56:32] | type=2
  i32x8 g1;
  g1[0] = 3 << 16;          // wg_mask=0, data_size=3 (8B)
  g1[1] = 8 << 16;          // tensor_dim0 = 8 elements/row (lo16 at bits 63:48)
  g1[2] = 0x2000 << 16;     // tensor_dim1 = 8192 rows (lo16 at bits 95:80)
  g1[3] = 8 << 16;          // tile_dim0 = 8 (bits 127:112); tensor_dim1 hi16 = 0
  g1[4] = 16;               // tile_dim1 = 16 valid indices; tile_dim2 = 0
  g1[5] = 8;                // tensor_dim0_stride = 8 elements (64 B records)
  g1[6] = 0;                // stride hi / dim1_stride (ignored in gather mode)
  g1[7] = 0;
  i32x4 g2 = { (int)idx[0], (int)idx[1], (int)idx[2], (int)idx[3] };
  i32x4 g3 = { (int)idx[4], (int)idx[5], (int)idx[6], (int)idx[7] };
#if __clang_major__ >= 23
  i32x8 gz = {0,0,0,0,0,0,0,0};
  __builtin_amdgcn_tensor_load_to_lds(g0, g1, g2, g3, gz, 0);
#else
  __builtin_amdgcn_tensor_load_to_lds(g0, g1, g2, g3, 0);
#endif
}

// Issue both halves of a 32-gaussian chunk. Tail indices are clamped to row 0
// (in-bounds; contribution later masked to zero via k<clen), keeping the real
// prefix ascending as required by gather-mode OOB rules.
__device__ __forceinline__ void tdm_issue_chunk(
    const float* __restrict__ grec, const unsigned short* q,
    unsigned c0, int clen, unsigned lds_off)
{
  const unsigned* qd = (const unsigned*)(q + c0);  // c0 % 32 == 0 -> dword aligned
  unsigned pr[16];
  #pragma unroll
  for (int i = 0; i < 16; ++i) {
    unsigned p = qd[i];
    if (2*i + 1 >= clen) p &= 0xffffu;
    if (2*i     >= clen) p  = 0u;
    pr[i] = (unsigned)__builtin_amdgcn_readfirstlane((int)p);
  }
  tdm_gather16(lds_off,         grec, pr);
  tdm_gather16(lds_off + 1024u, grec, pr + 8);
}
#endif // USE_TDM

__global__ __launch_bounds__(256) void render_kernel(
    const float* __restrict__ grec, float* __restrict__ out)
{
  __shared__ unsigned short        s_queue[GN];          // 16 KB compacted ids
  __shared__ __align__(16) float   s_rec[2 * 32 * REC_F];//  4 KB double-buffered
  __shared__ float                 s_stage[8 * 128];     //  4 KB D -> pixel stage
  __shared__ unsigned              s_wcnt[8];
  __shared__ unsigned              s_wmask[8];           // ballot fallback only

  const int tid  = threadIdx.x;
  const int lane = tid & 31;            // wave32
  const int wid  = tid >> 5;            // 8 waves
  const int tileX = blockIdx.x % TX_TILES;
  const int tileY = blockIdx.x / TX_TILES;
  const float txmin = (float)(tileX*TILE), txmax = txmin + (float)(TILE-1);
  const float tymin = (float)(tileY*TILE), tymax = tymin + (float)(TILE-1);

  // ---------- phase 1: scan all gaussians, compact overlaps (index order) ----------
  unsigned qlen = 0;
  for (int base = 0; base < GN; base += 256) {
    int gid = base + tid;
    if (base + 256 < GN)
      __builtin_prefetch(grec + (size_t)(base + 256 + tid) * REC_F, 0, 0);
    float4 bb = *(const float4*)(grec + (size_t)gid * REC_F);
    bool pred = (bb.x <= txmax) & (bb.y >= txmin) & (bb.z <= tymax) & (bb.w >= tymin);

#if __has_builtin(__builtin_amdgcn_ballot_w32)
    unsigned mask = __builtin_amdgcn_ballot_w32(pred);
#else
    if (lane == 0) s_wmask[wid] = 0u;
    __syncthreads();
    if (pred) atomicOr(&s_wmask[wid], 1u << lane);
    __syncthreads();
    unsigned mask = s_wmask[wid];
#endif
    if (lane == 0) s_wcnt[wid] = (unsigned)__popc(mask);
    __syncthreads();

    unsigned qbase = qlen, total = 0;
    #pragma unroll
    for (int w = 0; w < 8; ++w) {
      unsigned c = s_wcnt[w];
      total += c;
      if (w < wid) qbase += c;
    }
    unsigned pfx = (unsigned)__popc(mask & ((1u << lane) - 1u));
    if (pred) s_queue[qbase + pfx] = (unsigned short)gid;
    qlen += total;
    __syncthreads();
  }

  // ---------- phase 2: WMMA accumulation over 32-gaussian chunks ----------
  // Wave 'wid' owns image rows y0 = tileY*16 + 2*wid and y0+1.
  // A fragment (16x32 f16): lane L holds row M=L%16 (pixel x), K set
  //   {0..7,16..23} for L<16, {8..15,24..31} for L>=16 (ISA 7.12.2).
  // B fragment (32x16 f16): lane L holds col N=L%16, same K set as A.
  const int  mcol  = lane & 15;
  const bool lowh  = lane < 16;
  const int  kbase = lowh ? 0 : 8;
  const int  khalf = lowh ? 0 : 16;
  const float px  = txmin + (float)mcol;
  const float py0 = tymin + (float)(wid * 2);
  const float py1 = py0 + 1.f;

  v8f acc0 = {}; // rows y0: D[M=pixel x][N=channel rgbw]
  v8f acc1 = {}; // rows y1

  const unsigned nchunks = (qlen + 31u) >> 5;
#if USE_TDM
  const unsigned rec0_off = (unsigned)(unsigned long long)(const void*)&s_rec[0];
  // prologue: pull chunk 0 into buffer 0 via TDM
  if (wid == 0 && nchunks > 0) {
    int clen0 = (int)((qlen < 32u) ? qlen : 32u);
    tdm_issue_chunk(grec, s_queue, 0u, clen0, rec0_off);
    __builtin_amdgcn_s_wait_tensorcnt(0);
  }
  __syncthreads();
#endif

  for (unsigned ci = 0; ci < nchunks; ++ci) {
    const unsigned c0 = ci << 5;
    int clen = (int)(qlen - c0); if (clen > 32) clen = 32;
    const float* rec = s_rec + (ci & 1u) * (32 * REC_F);

#if USE_TDM
    // software pipeline: kick chunk ci+1 into the other buffer (TENSORcnt)
    if (wid == 0 && ci + 1 < nchunks) {
      unsigned c1 = c0 + 32u;
      int clenN = (int)(qlen - c1); if (clenN > 32) clenN = 32;
      tdm_issue_chunk(grec, s_queue, c1, clenN,
                      rec0_off + ((ci + 1u) & 1u) * (32u * REC_F * 4u));
    }
#else
    // synchronous fallback: vectorized float4 gather (128 lanes x 16B)
    {
      float* recw = (float*)rec;
      for (int t = tid; t < 32 * 4; t += 256) {
        int slot = t >> 2;
        int sl   = slot < clen ? slot : clen - 1;
        int id   = (int)s_queue[c0 + sl];
        float4 v = ((const float4*)(grec + (size_t)id * REC_F))[t & 3];
        ((float4*)recw)[t] = v;
      }
    }
    __syncthreads();
#endif

    v16h a0, a1, bf;
    #pragma unroll
    for (int j = 0; j < 16; ++j) {
      int k = kbase + (j < 8 ? j : j + 8);
      const float* rp = rec + k * REC_F;
      float xlo = rp[0], xhi = rp[1], ylo = rp[2], yhi = rp[3];
      float mux = rp[4], muy = rp[5], qa = rp[6], qb = rp[7];
      float qc  = rp[8], op  = rp[9];
      float dx  = px - mux;
      bool okx = (px >= xlo) & (px <= xhi) & (k < clen);

      float dy0 = py0 - muy;
      float e0  = (qa*dx + qb*dy0)*dx + qc*dy0*dy0;
      float w0  = op * __expf(-0.5f * e0);
      bool ok0  = okx & (py0 >= ylo) & (py0 <= yhi);
      a0[j] = (_Float16)(ok0 ? w0 : 0.f);

      float dy1 = py1 - muy;
      float e1  = (qa*dx + qb*dy1)*dx + qc*dy1*dy1;
      float w1  = op * __expf(-0.5f * e1);
      bool ok1  = okx & (py1 >= ylo) & (py1 <= yhi);
      a1[j] = (_Float16)(ok1 ? w1 : 0.f);
    }
    #pragma unroll
    for (int j = 0; j < 16; ++j) {
      float v = (mcol < 4) ? rec[(kbase + (j < 8 ? j : j + 8)) * REC_F + 10 + mcol] : 0.f;
      bf[j] = (_Float16)v;
    }

    // D = A(w) x B(rgbw) + C : v_wmma_f32_16x16x32_f16
    acc0 = __builtin_amdgcn_wmma_f32_16x16x32_f16(false, a0, false, bf,
                                                  (short)0, acc0, false, false);
    acc1 = __builtin_amdgcn_wmma_f32_16x16x32_f16(false, a1, false, bf,
                                                  (short)0, acc1, false, false);

#if USE_TDM
    if (wid == 0) __builtin_amdgcn_s_wait_tensorcnt(0); // chunk ci+1 landed
#endif
    __syncthreads();
  }

  // ---------- phase 3: normalize and write (each pixel exactly once) ----------
  // D layout: lane L, VGPR r -> (M = r + 8*(L>=16), N = L%16)
  #pragma unroll
  for (int r = 0; r < 8; ++r) {
    if (mcol < 4) {
      int M = r + (lowh ? 0 : 8);
      s_stage[wid*128 +      M*4 + mcol] = acc0[r];
      s_stage[wid*128 + 64 + M*4 + mcol] = acc1[r];
    }
  }
  __syncthreads();

  int pxl = tid & 15, pyl = tid >> 4;
  const float* st = s_stage + (pyl >> 1)*128 + (pyl & 1)*64 + pxl*4;
  float ww  = st[3];
  float inv = 1.f / (ww + EPS_C);
  size_t o = ((size_t)(tileY*TILE + pyl) * IMG_W + (size_t)(tileX*TILE + pxl)) * 3;
  out[o+0] = st[0] * inv;
  out[o+1] = st[1] * inv;
  out[o+2] = st[2] * inv;
}

extern "C" void kernel_launch(void* const* d_in, const int* in_sizes, int n_in,
                              void* d_out, int out_size, void* d_ws, size_t ws_size,
                              hipStream_t stream) {
  const float* xyz  = (const float*)d_in[0];
  const float* feat = (const float*)d_in[1];
  const float* opac = (const float*)d_in[2];
  const float* scal = (const float*)d_in[3];
  const float* rot  = (const float*)d_in[4];
  float* grec = (float*)d_ws;          // 8192 * 16 floats = 512 KB scratch
  float* out  = (float*)d_out;

  hipLaunchKernelGGL(prep_kernel, dim3(GN/256), dim3(256), 0, stream,
                     xyz, feat, opac, scal, rot, grec);
  hipLaunchKernelGGL(render_kernel, dim3(TX_TILES*TY_TILES), dim3(256), 0, stream,
                     grec, out);
  (void)in_sizes; (void)n_in; (void)out_size; (void)ws_size;
}
